// AlphaNet_v1_79362405695831
// MI455X (gfx1250) — compile-verified, matrix operand order checked by reading, not run on hardware
//
#include <hip/hip_runtime.h>
#include <hip/hip_bf16.h>

#define BATCH 131072
#define NN 9
#define TT 30
#define DD 10
#define NWIN 3
#define NROW 117            // 36 corr + 36 cov + 9*5
#define FLATN 702
#define FLATP 704           // padded K for WMMA (multiple of 32)
#define KSTEPS 22           // 704/32
#define EPS_BN 1e-5f
#define EPS 1e-8f

typedef __attribute__((ext_vector_type(16))) _Float16 v16h;
typedef __attribute__((ext_vector_type(8)))  float    v8f;

union AF { unsigned u[8]; v16h v; };
union HPK { _Float16 h[2]; unsigned u; };

__device__ __forceinline__ int row_group(int row) {
  return row < 36 ? 0 : row < 72 ? 1 : row < 81 ? 2 : row < 90 ? 3
       : row < 99 ? 4 : row < 108 ? 5 : 6;
}

// ---------------------------------------------------------------- K0: pack W1
// Layout: W1p[((kk*2+t)*32 + lane)*8 + v] = half2{ W1t[k][n], W1t[k+1][n] }
// with n = t*16 + (lane&15), k = kk*32 + (v<4 ? 2v : 16+2(v-4)) + 8*(lane>>4)
// (mirrors the 16-bit A-matrix 16x32 per-lane K pattern for the B operand)
__global__ void k_pack(const float* __restrict__ W1, unsigned* __restrict__ W1p) {
  int idx = blockIdx.x * 256 + threadIdx.x;
  if (idx >= KSTEPS * 2 * 32 * 8) return;
  int v    = idx & 7;
  int lane = (idx >> 3) & 31;
  int t    = (idx >> 8) & 1;
  int kk   = idx >> 9;
  int n    = t * 16 + (lane & 15);
  int half = lane >> 4;
  int kb   = kk * 32 + (v < 4 ? 2 * v : 16 + 2 * (v - 4)) + 8 * half;
  float f0 = (n < 30 && kb     < FLATN) ? W1[n * FLATN + kb]     : 0.f;
  float f1 = (n < 30 && kb + 1 < FLATN) ? W1[n * FLATN + kb + 1] : 0.f;
  HPK p; p.h[0] = (_Float16)f0; p.h[1] = (_Float16)f1;
  W1p[idx] = p.u;
}

// --------------------------------------------------------- K1: raw features
// one thread per sample; writes xraw[b][row][w] and per-block (sum,sum2) x7
__global__ void k_feats(const float* __restrict__ xb,
                        float* __restrict__ xraw,
                        float* __restrict__ partials) {
  int tid = threadIdx.x;
  long long b = (long long)blockIdx.x * 256 + tid;
  const float* xs = xb + b * (NN * TT);
  float* o = xraw + b * (NROW * NWIN);

  float ps[7]  = {0, 0, 0, 0, 0, 0, 0};
  float ps2[7] = {0, 0, 0, 0, 0, 0, 0};
#define ACC(gi, val) { float _v = (val); ps[gi] += _v; ps2[gi] += _v * _v; }

  const float invD = 1.f / (float)DD;
  const float inv9 = 1.f / (float)(DD - 1);
  const float invW = 1.f / 55.f;   // D*(D+1)/2

  for (int w = 0; w < NWIN; ++w) {
    float c[NN][DD];
    float sd[NN];
    for (int n = 0; n < NN; ++n) {
      float v[DD];
      float m = 0.f;
      for (int d = 0; d < DD; ++d) { v[d] = xs[n * TT + w * DD + d]; m += v[d]; }
      m *= invD;
      float s2 = 0.f, dec = 0.f;
      for (int d = 0; d < DD; ++d) {
        float cc = v[d] - m;
        c[n][d] = cc;
        s2 += cc * cc;
        dec += v[d] * ((float)(d + 1) * invW);
      }
      float st = sqrtf(s2 * inv9);
      sd[n] = st;
      float z  = m / (st + EPS);
      float rt = v[DD - 1] / v[0] - 1.f;
      o[(72 + n) * 3 + w] = st;  ACC(2, st);
      o[(81 + n) * 3 + w] = z;   ACC(3, z);
      o[(90 + n) * 3 + w] = rt;  ACC(4, rt);
      o[(99 + n) * 3 + w] = dec; ACC(5, dec);
      o[(108 + n) * 3 + w] = m;  ACC(6, m);
    }
    int p = 0;
    for (int i = 0; i < NN; ++i)
      for (int j = i + 1; j < NN; ++j, ++p) {
        float cv = 0.f;
        for (int d = 0; d < DD; ++d) cv += c[i][d] * c[j][d];
        cv *= inv9;
        float cr = cv / (sd[i] * sd[j] + EPS);
        o[p * 3 + w] = cr;        ACC(0, cr);
        o[(36 + p) * 3 + w] = cv; ACC(1, cv);
      }
  }
#undef ACC
  __shared__ float red[256];
  for (int q = 0; q < 14; ++q) {
    red[tid] = (q & 1) ? ps2[q >> 1] : ps[q >> 1];
    __syncthreads();
    for (int s = 128; s > 0; s >>= 1) {
      if (tid < s) red[tid] += red[tid + s];
      __syncthreads();
    }
    if (tid == 0) partials[blockIdx.x * 14 + q] = red[0];
    __syncthreads();
  }
}

// ------------------------------------------------- K2: finalize groups 0..6
__global__ void k_fin1(const float* __restrict__ partials, float* __restrict__ stats) {
  int tid = threadIdx.x;
  __shared__ float tmp[14];
  if (tid < 14) {
    float s = 0.f;
    for (int blk = 0; blk < 512; ++blk) s += partials[blk * 14 + tid];
    tmp[tid] = s;
  }
  __syncthreads();
  if (tid == 0) {
    for (int g = 0; g < 7; ++g) {
      float cnt  = (g < 2) ? (float)BATCH * 108.f : (float)BATCH * 27.f;
      float mean = tmp[2 * g] / cnt;
      float var  = tmp[2 * g + 1] / cnt - mean * mean;
      stats[2 * g]     = mean;
      stats[2 * g + 1] = rsqrtf(var + EPS_BN);
    }
  }
}

// ----------------------------------------- K3: pooled-feature batch stats
__global__ void k_pool_stats(const float* __restrict__ xraw,
                             const float* __restrict__ stats,
                             const float* __restrict__ gamma,
                             const float* __restrict__ beta,
                             float* __restrict__ partials2) {
  int tid = threadIdx.x;
  long long b = (long long)blockIdx.x * 256 + tid;
  __shared__ float sst[14];
  if (tid < 14) sst[tid] = stats[tid];
  __syncthreads();
  float g = gamma[0], be = beta[0];
  const float* r = xraw + b * (NROW * NWIN);
  float ps[3] = {0, 0, 0}, ps2[3] = {0, 0, 0};
  for (int row = 0; row < NROW; ++row) {
    int grp = row_group(row);
    float mu = sst[2 * grp], is = sst[2 * grp + 1];
    float n0 = g * (r[row * 3 + 0] - mu) * is + be;
    float n1 = g * (r[row * 3 + 1] - mu) * is + be;
    float n2 = g * (r[row * 3 + 2] - mu) * is + be;
    float mean = (n0 + n1 + n2) * (1.f / 3.f);
    float mx = fmaxf(n0, fmaxf(n1, n2));
    float mn = fminf(n0, fminf(n1, n2));
    ps[0] += mean; ps2[0] += mean * mean;
    ps[1] += mx;   ps2[1] += mx * mx;
    ps[2] += mn;   ps2[2] += mn * mn;
  }
  __shared__ float red[256];
  for (int q = 0; q < 6; ++q) {
    red[tid] = (q & 1) ? ps2[q >> 1] : ps[q >> 1];
    __syncthreads();
    for (int s = 128; s > 0; s >>= 1) {
      if (tid < s) red[tid] += red[tid + s];
      __syncthreads();
    }
    if (tid == 0) partials2[blockIdx.x * 6 + q] = red[0];
    __syncthreads();
  }
}

// ------------------------------------------------- K4: finalize groups 7..9
__global__ void k_fin2(const float* __restrict__ partials2, float* __restrict__ stats) {
  int tid = threadIdx.x;
  __shared__ float tmp[6];
  if (tid < 6) {
    float s = 0.f;
    for (int blk = 0; blk < 512; ++blk) s += partials2[blk * 6 + tid];
    tmp[tid] = s;
  }
  __syncthreads();
  if (tid == 0) {
    float cnt = (float)BATCH * 117.f;
    for (int g = 0; g < 3; ++g) {
      float mean = tmp[2 * g] / cnt;
      float var  = tmp[2 * g + 1] / cnt - mean * mean;
      stats[14 + 2 * g] = mean;
      stats[15 + 2 * g] = rsqrtf(var + EPS_BN);
    }
  }
}

// --------------------------------------------------------- K5: WMMA MLP
// 2 waves/block, each wave: build 16x704 f16 flat tile in LDS, then
// 22 k-steps x 2 n-tiles of v_wmma_f32_16x16x32_f16; ReLU+W2 in epilogue.
__global__ __launch_bounds__(64) void k_mlp(const float* __restrict__ xraw,
                                            const unsigned* __restrict__ W1p,
                                            const float* __restrict__ stats,
                                            const float* __restrict__ gamma,
                                            const float* __restrict__ beta,
                                            const float* __restrict__ b1,
                                            const float* __restrict__ W2,
                                            const float* __restrict__ b2,
                                            const float* __restrict__ w_out,
                                            const float* __restrict__ b_out,
                                            float* __restrict__ out) {
  __shared__ __align__(16) _Float16 flat[2][16][FLATP];
  __shared__ float sst[20];
  int tid  = threadIdx.x;
  int lane = tid & 31;
  int w    = tid >> 5;
  if (tid < 20) sst[tid] = stats[tid];
  __syncthreads();
  float g = gamma[0], be = beta[0];
  long long base = (long long)blockIdx.x * 32 + w * 16;

  // build normalized flat tiles (16 samples x 704 f16)
  for (int i = lane; i < 16 * NROW; i += 32) {
    int s = i / NROW, row = i - s * NROW;
    const float* r = xraw + (base + s) * (NROW * NWIN) + row * 3;
    int grp = row_group(row);
    float mu = sst[2 * grp], is = sst[2 * grp + 1];
    float n0 = g * (r[0] - mu) * is + be;
    float n1 = g * (r[1] - mu) * is + be;
    float n2 = g * (r[2] - mu) * is + be;
    flat[w][s][row * 3 + 0] = (_Float16)n0;
    flat[w][s][row * 3 + 1] = (_Float16)n1;
    flat[w][s][row * 3 + 2] = (_Float16)n2;
    float mean = (n0 + n1 + n2) * (1.f / 3.f);
    float mx = fmaxf(n0, fmaxf(n1, n2));
    float mn = fminf(n0, fminf(n1, n2));
    flat[w][s][351 + row] = (_Float16)(g * (mean - sst[14]) * sst[15] + be);
    flat[w][s][468 + row] = (_Float16)(g * (mx   - sst[16]) * sst[17] + be);
    flat[w][s][585 + row] = (_Float16)(g * (mn   - sst[18]) * sst[19] + be);
  }
  // zero K padding 702..703
  flat[w][lane & 15][702 + (lane >> 4)] = (_Float16)0.f;
  __syncthreads();

  int half = lane >> 4;
  const unsigned* rowA = (const unsigned*)&flat[w][lane & 15][0];
  int kidx[8];
#pragma unroll
  for (int v = 0; v < 8; ++v) kidx[v] = (v < 4 ? v : 4 + v) + 4 * half;

  v8f c0 = {}; v8f c1 = {};
  for (int kk = 0; kk < KSTEPS; ++kk) {
    AF a, bb0, bb1;
#pragma unroll
    for (int v = 0; v < 8; ++v) a.u[v] = rowA[kk * 16 + kidx[v]];
    const unsigned* bp = W1p + ((size_t)(kk * 2) * 32 + lane) * 8;
#pragma unroll
    for (int v = 0; v < 8; ++v) { bb0.u[v] = bp[v]; bb1.u[v] = bp[256 + v]; }
    c0 = __builtin_amdgcn_wmma_f32_16x16x32_f16(false, a.v, false, bb0.v,
                                                (short)0, c0, false, false);
    c1 = __builtin_amdgcn_wmma_f32_16x16x32_f16(false, a.v, false, bb1.v,
                                                (short)0, c1, false, false);
  }

  // epilogue: h = relu(acc + b1), y = (h . W2 + b2)*w_out + b_out
  int n0 = lane & 15, n1 = 16 + n0;
  float b1n0 = b1[n0],                  w2n0 = W2[n0];
  float b1n1 = (n1 < 30) ? b1[n1] : 0.f, w2n1 = (n1 < 30) ? W2[n1] : 0.f;
  float bb2 = b2[0], osc = w_out[0], ob = b_out[0];
#pragma unroll
  for (int rI = 0; rI < 8; ++rI) {
    float h0 = c0[rI] + b1n0; h0 = h0 > 0.f ? h0 : 0.f;
    float h1 = c1[rI] + b1n1; h1 = h1 > 0.f ? h1 : 0.f;
    float p = h0 * w2n0 + h1 * w2n1;
    p += __shfl_xor(p, 1);
    p += __shfl_xor(p, 2);
    p += __shfl_xor(p, 4);
    p += __shfl_xor(p, 8);
    if (n0 == 0) out[base + 8 * half + rI] = (p + bb2) * osc + ob;
  }
}

// ---------------------------------------------------------------- launcher
extern "C" void kernel_launch(void* const* d_in, const int* in_sizes, int n_in,
                              void* d_out, int out_size, void* d_ws, size_t ws_size,
                              hipStream_t stream) {
  const float* xb    = (const float*)d_in[0];
  const float* gamma = (const float*)d_in[1];
  const float* beta  = (const float*)d_in[2];
  const float* W1    = (const float*)d_in[3];
  const float* b1    = (const float*)d_in[4];
  const float* W2    = (const float*)d_in[5];
  const float* b2    = (const float*)d_in[6];
  const float* w_out = (const float*)d_in[7];
  const float* b_out = (const float*)d_in[8];
  float* out = (float*)d_out;

  char* ws = (char*)d_ws;
  size_t XRAW_OFF = 0;
  size_t xrawBytes = (size_t)BATCH * NROW * NWIN * sizeof(float); // 184 MB
  size_t P1_OFF = XRAW_OFF + xrawBytes;        // 512*14 floats
  size_t P2_OFF = P1_OFF + 512 * 14 * 4;       // 512*6 floats
  size_t ST_OFF = P2_OFF + 512 * 6 * 4;        // 20 floats (padded)
  size_t W1P_OFF = ST_OFF + 128;               // 11264 dwords

  float*    xraw  = (float*)(ws + XRAW_OFF);
  float*    p1    = (float*)(ws + P1_OFF);
  float*    p2    = (float*)(ws + P2_OFF);
  float*    stats = (float*)(ws + ST_OFF);
  unsigned* W1p   = (unsigned*)(ws + W1P_OFF);

  k_pack<<<44, 256, 0, stream>>>(W1, W1p);
  k_feats<<<BATCH / 256, 256, 0, stream>>>(xb, xraw, p1);
  k_fin1<<<1, 64, 0, stream>>>(p1, stats);
  k_pool_stats<<<BATCH / 256, 256, 0, stream>>>(xraw, stats, gamma, beta, p2);
  k_fin2<<<1, 64, 0, stream>>>(p2, stats);
  k_mlp<<<BATCH / 32, 64, 0, stream>>>(xraw, W1p, stats, gamma, beta,
                                       b1, W2, b2, w_out, b_out, out);
}
